// RoSEMultiheadSelfAttention_51685636440494
// MI455X (gfx1250) — matrix operand order repaired and checked by reading, not compiled, not run on hardware
//
#include <hip/hip_runtime.h>
#include <hip/hip_bf16.h>

#define DIMX 768
#define HEADS 12
#define HEAD_DIM 64
#define DHALF 32
#define NB 2048
#define BATCH 2
#define M_ROWS (BATCH * NB)   // 4096
#define E_QKV (3 * DIMX)      // 2304

typedef __attribute__((ext_vector_type(16))) __bf16 v16bf;
typedef __attribute__((ext_vector_type(8)))  __bf16 v8bf;
typedef __attribute__((ext_vector_type(8)))  float  v8f;

// ---- WMMA fragment loaders (bf16, wave32, 16x16x32) ----------------------
// A 16x32 (MxK) from row-major [M, ld]: lane<16 -> row=lane, K = 0..7 & 16..23
//                                       lane>=16 -> row=lane-16, K = 8..15 & 24..31
__device__ __forceinline__ v16bf load_a16x32(const __bf16* p, int ld) {
  int lane = threadIdx.x & 31;
  int m = lane & 15;
  int h8 = (lane >> 4) << 3;
  const __bf16* r = p + (size_t)m * ld + h8;
  union { v16bf v; v8bf h[2]; } u;
  u.h[0] = *(const v8bf*)(r);
  u.h[1] = *(const v8bf*)(r + 16);
  return u.v;
}

// B 32x16 (KxN) where element (k,n) comes from p[n*ld + k]  (source is
// row-major [N, ld] -- "transposed" operand; lane<16: K = 0..7 & 16..23,
// lane>=16: K = 8..15 & 24..31).
__device__ __forceinline__ v16bf load_bT32x16(const __bf16* p, int ld) {
  int lane = threadIdx.x & 31;
  int n = lane & 15;
  int kh = (lane >> 4) << 3;
  const __bf16* r = p + (size_t)n * ld + kh;
  union { v16bf v; v8bf h[2]; } u;
  u.h[0] = *(const v8bf*)(r);
  u.h[1] = *(const v8bf*)(r + 16);
  return u.v;
}

__device__ __forceinline__ v8f wmma_bf16(v16bf a, v16bf b, v8f c) {
  return __builtin_amdgcn_wmma_f32_16x16x32_bf16(false, a, false, b, (short)0, c,
                                                 false, false);
}

// ---- fp32 -> bf16 conversion ---------------------------------------------
__global__ void cvt_f32_bf16(const float* __restrict__ s, __bf16* __restrict__ d,
                             int n) {
  int i = blockIdx.x * blockDim.x + threadIdx.x;
  if (i < n) d[i] = (__bf16)s[i];
}

// ---- QKV projection: qkv[b,n,e] = sum_d x[b,n,d] * w_qkv[e,d] ------------
// 2x2 register-blocked: each wave computes a 32x32 output block.
// Scatters into q[b,h,n,d], k[b,h,n,d] and TRANSPOSED v[b,h,d,n].
__device__ __forceinline__ void scatter_qkv(v8f c, int r0, int e0, int lane,
                                            __bf16* __restrict__ qbf,
                                            __bf16* __restrict__ kbf,
                                            __bf16* __restrict__ vtbf) {
  int half = lane >> 4;
  int e = e0 + (lane & 15);
  int which = e / DIMX;
  int rme = e % DIMX;
  int h = rme / HEAD_DIM, d = rme % HEAD_DIM;
#pragma unroll
  for (int i = 0; i < 8; ++i) {
    int row = r0 + i + 8 * half;
    int b_ = row >> 11, n = row & (NB - 1);
    __bf16 val = (__bf16)c[i];
    if (which == 0)
      qbf[(((size_t)(b_ * HEADS + h) * NB) + n) * HEAD_DIM + d] = val;
    else if (which == 1)
      kbf[(((size_t)(b_ * HEADS + h) * NB) + n) * HEAD_DIM + d] = val;
    else
      vtbf[(((size_t)(b_ * HEADS + h) * HEAD_DIM) + d) * NB + n] = val;
  }
}

__global__ void __launch_bounds__(128) qkv_gemm(
    const __bf16* __restrict__ xbf, const __bf16* __restrict__ wbf,
    __bf16* __restrict__ qbf, __bf16* __restrict__ kbf, __bf16* __restrict__ vtbf) {
  int wid = threadIdx.x >> 5;
  int e0 = blockIdx.x * 128 + wid * 32;
  int r0 = blockIdx.y * 32;
  v8f c00 = {}, c01 = {}, c10 = {}, c11 = {};
  for (int k = 0; k < DIMX; k += 32) {
    v16bf a0 = load_a16x32(xbf + (size_t)r0 * DIMX + k, DIMX);
    v16bf a1 = load_a16x32(xbf + (size_t)(r0 + 16) * DIMX + k, DIMX);
    v16bf b0 = load_bT32x16(wbf + (size_t)e0 * DIMX + k, DIMX);
    v16bf b1 = load_bT32x16(wbf + (size_t)(e0 + 16) * DIMX + k, DIMX);
    c00 = wmma_bf16(a0, b0, c00);
    c01 = wmma_bf16(a0, b1, c01);
    c10 = wmma_bf16(a1, b0, c10);
    c11 = wmma_bf16(a1, b1, c11);
  }
  int lane = threadIdx.x & 31;
  scatter_qkv(c00, r0, e0, lane, qbf, kbf, vtbf);
  scatter_qkv(c01, r0, e0 + 16, lane, qbf, kbf, vtbf);
  scatter_qkv(c10, r0 + 16, e0, lane, qbf, kbf, vtbf);
  scatter_qkv(c11, r0 + 16, e0 + 16, lane, qbf, kbf, vtbf);
}

// ---- RoPE on q and k in place --------------------------------------------
__global__ void rope_kernel(__bf16* __restrict__ q, __bf16* __restrict__ k,
                            const float* __restrict__ theta,
                            const float* __restrict__ voxel,
                            const int* __restrict__ pgd,
                            const int* __restrict__ pgh,
                            const int* __restrict__ pgw) {
  (void)pgd;
  int idx = blockIdx.x * blockDim.x + threadIdx.x;   // (bh, n, d)
  int d = idx & (DHALF - 1);
  int t = idx >> 5;
  int n = t & (NB - 1);
  int bh = t >> 11;
  int h = bh % HEADS;
  int gh = *pgh, gw = *pgw;
  int hw = gh * gw;
  int iz = n / hw;
  int rem = n - iz * hw;
  int iy = rem / gw;
  int ix = rem - iy * gw;
  int axis = d % 3;
  float coord = (axis == 0) ? (float)iz * voxel[0]
              : (axis == 1) ? (float)iy * voxel[1]
                            : (float)ix * voxel[2];
  float ang = coord * theta[h * DHALF + d];
  float sn, cs;
  __sincosf(ang, &sn, &cs);
  size_t base = ((size_t)bh * NB + n) * HEAD_DIM + d;
  float q1 = (float)q[base], q2 = (float)q[base + DHALF];
  q[base]         = (__bf16)(q1 * cs - q2 * sn);
  q[base + DHALF] = (__bf16)(q1 * sn + q2 * cs);
  float k1 = (float)k[base], k2 = (float)k[base + DHALF];
  k[base]         = (__bf16)(k1 * cs - k2 * sn);
  k[base + DHALF] = (__bf16)(k1 * sn + k2 * cs);
}

// ---- Flash attention: one wave = 32 query rows (2 row-tiles) --------------
// K/V B-fragments are loaded once and shared by both row-tiles.
__global__ void __launch_bounds__(128) attn_kernel(
    const __bf16* __restrict__ qbf, const __bf16* __restrict__ kbf,
    const __bf16* __restrict__ vtbf, __bf16* __restrict__ obf) {
  __shared__ __align__(32) __bf16 plds[4][2][16 * 32];
  int wid = threadIdx.x >> 5;
  int lane = threadIdx.x & 31;
  int half = lane >> 4;
  int b = blockIdx.z, h = blockIdx.y;
  int q0 = blockIdx.x * 128 + wid * 32;
  const __bf16* Q  = qbf  + ((size_t)(b * HEADS + h) * NB) * HEAD_DIM;
  const __bf16* K  = kbf  + ((size_t)(b * HEADS + h) * NB) * HEAD_DIM;
  const __bf16* VT = vtbf + ((size_t)(b * HEADS + h) * HEAD_DIM) * NB;

  v16bf qa[2][2];
#pragma unroll
  for (int rt = 0; rt < 2; ++rt) {
    qa[rt][0] = load_a16x32(Q + (size_t)(q0 + rt * 16) * HEAD_DIM, HEAD_DIM);
    qa[rt][1] = load_a16x32(Q + (size_t)(q0 + rt * 16) * HEAD_DIM + 32, HEAD_DIM);
  }
  v8f zero = {};
  v8f o[2][4];
  float mx[2][8], sm[2][8];
#pragma unroll
  for (int rt = 0; rt < 2; ++rt) {
#pragma unroll
    for (int g = 0; g < 4; ++g) o[rt][g] = zero;
#pragma unroll
    for (int i = 0; i < 8; ++i) { mx[rt][i] = -1e30f; sm[rt][i] = 0.f; }
  }
  const float scale = 0.125f;  // 1/sqrt(64)

  for (int j = 0; j < NB; j += 32) {
    // Prefetch next K/V 32-key block into cache (global_prefetch_b8).
    __builtin_prefetch((const void*)(K + (size_t)(j + 32) * HEAD_DIM +
                                     (size_t)lane * 64), 0, 0);
    __builtin_prefetch((const void*)(VT + (size_t)(lane * 2) * NB + j + 32), 0, 0);

    // S = Q K^T : two 16x16 key tiles, shared K fragments across row tiles.
    v8f s[2][2] = {{zero, zero}, {zero, zero}};
#pragma unroll
    for (int kt = 0; kt < 2; ++kt) {
#pragma unroll
      for (int kk = 0; kk < 2; ++kk) {
        v16bf bt = load_bT32x16(
            K + (size_t)(j + kt * 16) * HEAD_DIM + kk * 32, HEAD_DIM);
        s[0][kt] = wmma_bf16(qa[0][kk], bt, s[0][kt]);
        s[1][kt] = wmma_bf16(qa[1][kk], bt, s[1][kt]);
      }
    }
    // Online softmax per row tile; P -> LDS (C-layout -> row-major 16x32).
#pragma unroll
    for (int rt = 0; rt < 2; ++rt) {
      __bf16* myp = &plds[wid][rt][0];
      float f[8];
#pragma unroll
      for (int i = 0; i < 8; ++i) {
        float a0 = s[rt][0][i] * scale, a1 = s[rt][1][i] * scale;
        float m = fmaxf(a0, a1);
        m = fmaxf(m, __shfl_xor(m, 1, 32));
        m = fmaxf(m, __shfl_xor(m, 2, 32));
        m = fmaxf(m, __shfl_xor(m, 4, 32));
        m = fmaxf(m, __shfl_xor(m, 8, 32));
        float nm = fmaxf(mx[rt][i], m);
        f[i] = __expf(mx[rt][i] - nm);
        mx[rt][i] = nm;
        float p0 = __expf(a0 - nm), p1 = __expf(a1 - nm);
        float ps = p0 + p1;
        ps += __shfl_xor(ps, 1, 32);
        ps += __shfl_xor(ps, 2, 32);
        ps += __shfl_xor(ps, 4, 32);
        ps += __shfl_xor(ps, 8, 32);
        sm[rt][i] = sm[rt][i] * f[i] + ps;
        int row = i + 8 * half, col = lane & 15;
        myp[row * 32 + col]      = (__bf16)p0;
        myp[row * 32 + 16 + col] = (__bf16)p1;
        o[rt][0][i] *= f[i]; o[rt][1][i] *= f[i];
        o[rt][2][i] *= f[i]; o[rt][3][i] *= f[i];
      }
    }
    asm volatile("s_wait_dscnt 0" ::: "memory");
    v16bf pa0 = load_a16x32(&plds[wid][0][0], 32);
    v16bf pa1 = load_a16x32(&plds[wid][1][0], 32);
    // O += P * V : V fragment loaded once per column group, shared by both
    // row tiles (V^T layout [d, keys] -> contiguous B loads).
#pragma unroll
    for (int g = 0; g < 4; ++g) {
      v16bf vb = load_bT32x16(VT + (size_t)(g * 16) * NB + j, NB);
      o[0][g] = wmma_bf16(pa0, vb, o[0][g]);
      o[1][g] = wmma_bf16(pa1, vb, o[1][g]);
    }
  }
#pragma unroll
  for (int rt = 0; rt < 2; ++rt) {
#pragma unroll
    for (int i = 0; i < 8; ++i) {
      float inv = 1.f / sm[rt][i];
      int n = q0 + rt * 16 + i + 8 * half;
      size_t rb = ((size_t)(b * NB + n)) * DIMX + h * HEAD_DIM + (lane & 15);
      obf[rb +  0] = (__bf16)(o[rt][0][i] * inv);
      obf[rb + 16] = (__bf16)(o[rt][1][i] * inv);
      obf[rb + 32] = (__bf16)(o[rt][2][i] * inv);
      obf[rb + 48] = (__bf16)(o[rt][3][i] * inv);
    }
  }
}

// ---- Output projection + bias (2x2 register-blocked) ----------------------
__global__ void __launch_bounds__(128) proj_gemm(
    const __bf16* __restrict__ abf, const __bf16* __restrict__ wbf,
    const float* __restrict__ bias, float* __restrict__ out) {
  int wid = threadIdx.x >> 5;
  int e0 = blockIdx.x * 128 + wid * 32;
  int r0 = blockIdx.y * 32;
  v8f c00 = {}, c01 = {}, c10 = {}, c11 = {};
  for (int k = 0; k < DIMX; k += 32) {
    v16bf a0 = load_a16x32(abf + (size_t)r0 * DIMX + k, DIMX);
    v16bf a1 = load_a16x32(abf + (size_t)(r0 + 16) * DIMX + k, DIMX);
    v16bf b0 = load_bT32x16(wbf + (size_t)e0 * DIMX + k, DIMX);
    v16bf b1 = load_bT32x16(wbf + (size_t)(e0 + 16) * DIMX + k, DIMX);
    c00 = wmma_bf16(a0, b0, c00);
    c01 = wmma_bf16(a0, b1, c01);
    c10 = wmma_bf16(a1, b0, c10);
    c11 = wmma_bf16(a1, b1, c11);
  }
  int lane = threadIdx.x & 31;
  int half = lane >> 4;
  int eA = e0 + (lane & 15);
  int eB = eA + 16;
  float bvA = bias[eA], bvB = bias[eB];
#pragma unroll
  for (int i = 0; i < 8; ++i) {
    int rowA = r0 + i + 8 * half;
    int rowB = rowA + 16;
    out[(size_t)rowA * DIMX + eA] = c00[i] + bvA;
    out[(size_t)rowA * DIMX + eB] = c01[i] + bvB;
    out[(size_t)rowB * DIMX + eA] = c10[i] + bvA;
    out[(size_t)rowB * DIMX + eB] = c11[i] + bvB;
  }
}

extern "C" void kernel_launch(void* const* d_in, const int* in_sizes, int n_in,
                              void* d_out, int out_size, void* d_ws, size_t ws_size,
                              hipStream_t stream) {
  (void)in_sizes; (void)n_in; (void)out_size; (void)ws_size;
  const float* x      = (const float*)d_in[0];
  const float* voxel  = (const float*)d_in[1];
  const float* w_qkv  = (const float*)d_in[2];
  const float* w_proj = (const float*)d_in[3];
  const float* b_proj = (const float*)d_in[4];
  const float* theta  = (const float*)d_in[5];
  const int*   gd     = (const int*)d_in[6];
  const int*   gh     = (const int*)d_in[7];
  const int*   gw     = (const int*)d_in[8];

  __bf16* ws = (__bf16*)d_ws;
  size_t off = 0;
  __bf16* xbf     = ws + off; off += (size_t)M_ROWS * DIMX;
  __bf16* wqkvbf  = ws + off; off += (size_t)E_QKV * DIMX;
  __bf16* wprojbf = ws + off; off += (size_t)DIMX * DIMX;
  __bf16* qbf     = ws + off; off += (size_t)BATCH * HEADS * NB * HEAD_DIM;
  __bf16* kbf     = ws + off; off += (size_t)BATCH * HEADS * NB * HEAD_DIM;
  __bf16* vtbf    = ws + off; off += (size_t)BATCH * HEADS * NB * HEAD_DIM;
  __bf16* attnbf  = ws + off; off += (size_t)M_ROWS * DIMX;

  int nx = M_ROWS * DIMX, nw = E_QKV * DIMX, np = DIMX * DIMX;
  cvt_f32_bf16<<<(nx + 255) / 256, 256, 0, stream>>>(x, xbf, nx);
  cvt_f32_bf16<<<(nw + 255) / 256, 256, 0, stream>>>(w_qkv, wqkvbf, nw);
  cvt_f32_bf16<<<(np + 255) / 256, 256, 0, stream>>>(w_proj, wprojbf, np);

  qkv_gemm<<<dim3(E_QKV / 128, M_ROWS / 32), 128, 0, stream>>>(xbf, wqkvbf, qbf,
                                                               kbf, vtbf);

  int nrope = BATCH * HEADS * NB * DHALF;
  rope_kernel<<<nrope / 256, 256, 0, stream>>>(qbf, kbf, theta, voxel, gd, gh, gw);

  attn_kernel<<<dim3(NB / 128, HEADS, BATCH), 128, 0, stream>>>(qbf, kbf, vtbf,
                                                                attnbf);

  proj_gemm<<<dim3(DIMX / 128, M_ROWS / 32), 128, 0, stream>>>(attnbf, wprojbf,
                                                               b_proj,
                                                               (float*)d_out);
}
